// TestBinary4_90434831385370
// MI455X (gfx1250) — matrix operand order repaired and checked by reading, not compiled, not run on hardware
//
#include <hip/hip_runtime.h>
#include <hip/hip_bf16.h>
#include <stdint.h>

// MI455X / gfx1250, wave32. D = A(bf16) x B(bf16) + C(f32) via v_wmma_f32_16x16x32_bf16.
typedef __attribute__((ext_vector_type(16))) __bf16 v16bf;
typedef __attribute__((ext_vector_type(8)))  float  v8f;
typedef __attribute__((ext_vector_type(4)))  float  v4f;
typedef __attribute__((ext_vector_type(4)))  int    v4i;
typedef __attribute__((ext_vector_type(4)))  unsigned int v4u;

// CDNA5 async global->LDS copy (ASYNCcnt-tracked, no VGPR staging).
#if defined(__has_builtin)
#if __has_builtin(__builtin_amdgcn_global_load_async_to_lds_b128)
#define ASYNC_LDS 1
#endif
#endif
#ifndef ASYNC_LDS
#define ASYNC_LDS 0
#endif

#if ASYNC_LDS
typedef __attribute__((address_space(1))) v4i* gp_v4i;   // global ptr to <4 x i32>
typedef __attribute__((address_space(3))) v4i* lp_v4i;   // LDS ptr to <4 x i32>
__device__ __forceinline__ void async_copy16(const v4u* g, v4u* l) {
  // Generic->AS casts via integer: global flat address is identity;
  // LDS byte offset = low 32 bits of the generic shared address.
  __builtin_amdgcn_global_load_async_to_lds_b128(
      (gp_v4i)(uintptr_t)g, (lp_v4i)(uint32_t)(uintptr_t)l, 0, 0);
}
#endif

// Problem: Y[8192 x 4096] = Xr[8192 x 512] @ Wu[512 x 4096]
//   Xr[m,kk] = sum_{r<8} x[m, 512*r + kk]      (weight matrix repeats every 512 rows)
//   Wu[kk,n] = (float)kernel[kk*4096 + n], values in {-1,0,1} (exact in bf16)
// Precision: Xr split into bf16 hi + bf16 lo; both matmul'd vs Wu into one f32 accum.

// ---------------------------------------------------------------------------
// Phase 1: fold X (8192x4096 f32) -> Xr (8192x512), split hi/lo, store in
// WMMA A-fragment order:  [mTile(128)][kStep(16)][mSub(4)][lane(32)][half(2)][8 bf16]
// A 16x32 bf16 order used here: lane<16 (M=lane): elems K0..15;
// lane>=16 (M=lane-16): elems K16..31.
// ---------------------------------------------------------------------------
__global__ __launch_bounds__(256) void fold_split_kernel(
    const float* __restrict__ x, __bf16* __restrict__ ahi, __bf16* __restrict__ alo) {
  const int t  = blockIdx.x * 256 + threadIdx.x;   // 524288 threads total
  const int m  = t >> 6;                            // 0..8191
  const int kg = t & 63;                            // group of 8 consecutive kk
  const float* xp = x + (size_t)m * 4096 + kg * 8;

  float acc[8];
#pragma unroll
  for (int i = 0; i < 8; ++i) acc[i] = 0.0f;
#pragma unroll
  for (int r = 0; r < 8; ++r) {
    // X is read exactly once across the whole problem: nontemporal (TH=NT)
    // so it does not evict the reused A/W working set from L2.
    const v4f a = __builtin_nontemporal_load((const v4f*)(xp + (size_t)r * 512));
    const v4f b = __builtin_nontemporal_load((const v4f*)(xp + (size_t)r * 512) + 1);
#pragma unroll
    for (int i = 0; i < 4; ++i) { acc[i] += a[i]; acc[4 + i] += b[i]; }
  }

  const int mTile = m >> 6, mSub = (m >> 4) & 3, mIn = m & 15;
  const int kStep = kg >> 2;          // k/32
  const int q     = kg & 3;           // (k%32)/8
  const int lane  = mIn + 16 * (q >> 1);
  const int half  = q & 1;
  const size_t base =
      ((((size_t)(mTile * 16 + kStep)) * 4 + mSub) * 32 + lane) * 16 + half * 8;

  union { __bf16 h[8]; v4u v; } hi, lo;
#pragma unroll
  for (int i = 0; i < 8; ++i) {
    float  f = acc[i];
    __bf16 h = (__bf16)f;             // RNE
    hi.h[i]  = h;
    lo.h[i]  = (__bf16)(f - (float)h);
  }
  *(v4u*)(ahi + base) = hi.v;
  *(v4u*)(alo + base) = lo.v;
}

// ---------------------------------------------------------------------------
// Phase 1b: expand kernel (int) -> bf16 in WMMA B-fragment order:
//   [nTile(32)][kStep(16)][nSub(8)][lane(32)][16 bf16]
// B 32x16 bf16 order used here (same as A): lane<16 (N=lane): elems K0..15; lane>=16: elems K16..31.
// ---------------------------------------------------------------------------
__global__ __launch_bounds__(256) void wexpand_kernel(
    const int* __restrict__ kern, __bf16* __restrict__ wf) {
  const int t  = blockIdx.x * 256 + threadIdx.x;   // 131072 threads
  const int kg = t >> 12;                           // 0..31 : group of 16 k
  const int n  = t & 4095;

  union { __bf16 h[16]; v4u v[2]; } w;
#pragma unroll
  for (int i = 0; i < 16; ++i) {
    const int k = kg * 16 + i;
    w.h[i] = (__bf16)(float)kern[(size_t)k * 4096 + n];  // {-1,0,1}: exact
  }
  const int nTile = n >> 7, nSub = (n >> 4) & 7;
  const int kStep = kg >> 1;
  const int lane  = (n & 15) + 16 * (kg & 1);
  const size_t base =
      ((((size_t)(nTile * 16 + kStep)) * 8 + nSub) * 32 + lane) * 16;
  *(v4u*)(wf + base)     = w.v[0];
  *(v4u*)(wf + base + 8) = w.v[1];
}

// ---------------------------------------------------------------------------
// Phase 2: GEMM. Block = 256 thr (8 waves), tile 64(M) x 128(N), K chunk 64.
// Waves 4(M) x 2(N); each wave: 4 subtiles of 16x16, hi+lo WMMA per subtile.
// LDS = 8KB Ahi + 8KB Alo + 16KB B = 32KB, all fragment-order contiguous copies.
// ---------------------------------------------------------------------------
__global__ __launch_bounds__(256) void gemm_kernel(
    const __bf16* __restrict__ ahi, const __bf16* __restrict__ alo,
    const __bf16* __restrict__ wf, float* __restrict__ out) {
  __shared__ v16bf sAhi[8][32];   // [kStep*4+mSub][lane]
  __shared__ v16bf sAlo[8][32];
  __shared__ v16bf sB[16][32];    // [kStep*8+nSub][lane]

  const int tid   = threadIdx.x;
  const int nTile = blockIdx.x;   // 0..31
  const int mTile = blockIdx.y;   // 0..127
  const int wave  = tid >> 5, lane = tid & 31;
  const int waveM = wave & 3, waveN = wave >> 2;

  v8f c0 = {0.f,0.f,0.f,0.f,0.f,0.f,0.f,0.f};
  v8f c[4] = {c0, c0, c0, c0};

  for (int ch = 0; ch < 8; ++ch) {
    const v4u* gAhi = (const v4u*)(ahi + ((size_t)(mTile * 16 + ch * 2)) * 2048);
    const v4u* gAlo = (const v4u*)(alo + ((size_t)(mTile * 16 + ch * 2)) * 2048);
    const v4u* gB   = (const v4u*)(wf  + ((size_t)(nTile * 16 + ch * 2)) * 4096);
    v4u* lAhi = (v4u*)sAhi;   // 512 x 16B
    v4u* lAlo = (v4u*)sAlo;   // 512 x 16B
    v4u* lB   = (v4u*)sB;     // 1024 x 16B
#if ASYNC_LDS
    async_copy16(gAhi + tid,       lAhi + tid);
    async_copy16(gAhi + tid + 256, lAhi + tid + 256);
    async_copy16(gAlo + tid,       lAlo + tid);
    async_copy16(gAlo + tid + 256, lAlo + tid + 256);
#pragma unroll
    for (int j = 0; j < 4; ++j) async_copy16(gB + tid + 256 * j, lB + tid + 256 * j);
    asm volatile("s_wait_asynccnt 0x0" ::: "memory");
#else
    lAhi[tid]       = gAhi[tid];
    lAhi[tid + 256] = gAhi[tid + 256];
    lAlo[tid]       = gAlo[tid];
    lAlo[tid + 256] = gAlo[tid + 256];
#pragma unroll
    for (int j = 0; j < 4; ++j) lB[tid + 256 * j] = gB[tid + 256 * j];
#endif
    __syncthreads();

#pragma unroll
    for (int ks = 0; ks < 2; ++ks) {
      const v16bf aH = sAhi[ks * 4 + waveM][lane];
      const v16bf aL = sAlo[ks * 4 + waveM][lane];
#pragma unroll
      for (int ns = 0; ns < 4; ++ns) {
        const v16bf b = sB[ks * 8 + waveN * 4 + ns][lane];
        c[ns] = __builtin_amdgcn_wmma_f32_16x16x32_bf16(
            false, aH, false, b, (short)0, c[ns], false, false);
        c[ns] = __builtin_amdgcn_wmma_f32_16x16x32_bf16(
            false, aL, false, b, (short)0, c[ns], false, false);
      }
    }
    __syncthreads();
  }

  // C/D 16x16 f32: VGPR r -> M = r + 8*(lane>=16), N = lane%16.
  // Y is written once and never re-read: nontemporal stores keep L2 for A/W.
  const int row0 = mTile * 64 + waveM * 16 + 8 * (lane >> 4);
  const int col0 = nTile * 128 + waveN * 64 + (lane & 15);
#pragma unroll
  for (int ns = 0; ns < 4; ++ns) {
#pragma unroll
    for (int r = 0; r < 8; ++r) {
      __builtin_nontemporal_store(
          c[ns][r], &out[(size_t)(row0 + r) * 4096 + (col0 + ns * 16)]);
    }
  }
}

// ---------------------------------------------------------------------------
extern "C" void kernel_launch(void* const* d_in, const int* in_sizes, int n_in,
                              void* d_out, int out_size, void* d_ws, size_t ws_size,
                              hipStream_t stream) {
  const float* x    = (const float*)d_in[0];   // (2,4096,4096) f32
  const int*   kern = (const int*)d_in[1];     // 2097152 ints in {-1,0,1}
  float* out = (float*)d_out;                  // (2,4096,4096) f32

  // Workspace layout (20 MiB): Ahi 8 MiB | Alo 8 MiB | Wfrag 4 MiB
  char* ws = (char*)d_ws;
  __bf16* ahi = (__bf16*)(ws);
  __bf16* alo = (__bf16*)(ws + ((size_t)8 << 20));
  __bf16* wf  = (__bf16*)(ws + ((size_t)16 << 20));

  fold_split_kernel<<<2048, 256, 0, stream>>>(x, ahi, alo);
  wexpand_kernel<<<512, 256, 0, stream>>>(kern, wf);
  gemm_kernel<<<dim3(32, 128), 256, 0, stream>>>(ahi, alo, wf, out);
}
